// GraphEncoderTL_6751688589919
// MI455X (gfx1250) — matrix operand order repaired and checked by reading, not compiled, not run on hardware
//
#include <hip/hip_runtime.h>
#include <hip/hip_bf16.h>

// ---------------------------------------------------------------------------
// GraphEncoderTL for MI455X (gfx1250, wave32, WMMA)
//  - All dense GEMMs: v_wmma_f32_16x16x32_bf16, operands pre-packed to padded
//    bf16 (A row-major [Mp,Kp], B transposed [Np,Kp]) so the inner loop is
//    4x clause-grouped b128 loads + 1 WMMA; latency hidden by occupancy.
//  - Sparse GAT: wave-per-edge scatter with native f32 global atomics.
// ---------------------------------------------------------------------------

typedef __attribute__((ext_vector_type(16))) __bf16 v16bf;
typedef __attribute__((ext_vector_type(8)))  __bf16 v8bf;
typedef __attribute__((ext_vector_type(8)))  float  v8f;

__device__ __forceinline__ __bf16 f2bf(float f) {
  unsigned u = __builtin_bit_cast(unsigned, f);
  unsigned r = (u + 0x7fffu + ((u >> 16) & 1u)) >> 16;   // round-to-nearest-even
  unsigned short s = (unsigned short)r;
  return __builtin_bit_cast(__bf16, s);
}

__device__ __forceinline__ void atomicMaxF32(float* addr, float val) {
  unsigned* ua = (unsigned*)addr;
  unsigned old = *ua;
  while (__uint_as_float(old) < val) {
    unsigned assumed = old;
    old = atomicCAS(ua, assumed, __float_as_uint(val));
    if (old == assumed) break;
  }
}

// ---------------------------------------------------------------------------
// WMMA GEMM on pre-packed bf16: C[Mp,Np] (+)= A[Mp,Kp] * BT[Np,Kp]^T
// One wave per 16x16 tile; blockIdx.y = split-K slice (f32 atomic add).
// Mp,Np multiples of 16; Kp multiple of 32; buffers zero-padded.
// ---------------------------------------------------------------------------
__global__ void k_wmma_gemm(const __bf16* __restrict__ A, const __bf16* __restrict__ BT,
                            float* __restrict__ C, const float* __restrict__ bias,
                            int Mp, int Np, int Kp, int lda, int ldb, int ldc,
                            int splitK)
{
  int wave = blockIdx.x * (blockDim.x >> 5) + (threadIdx.x >> 5);
  int tilesN = Np >> 4;
  if (wave >= (Mp >> 4) * tilesN) return;          // wave-uniform: EXEC stays full
  int tm = wave / tilesN, tn = wave % tilesN;

  int lane = threadIdx.x & 31;
  int half = lane >> 4;
  int l16  = lane & 15;

  int kchunk = (((Kp + splitK - 1) / splitK) + 31) & ~31;
  int k0   = blockIdx.y * kchunk;
  int kend = min(Kp, k0 + kchunk);

  // A fragment (16x32 bf16): row m=tm*16+l16; K runs [half*8..+7] and [16+half*8..+7]
  const __bf16* arow = A  + (size_t)(tm * 16 + l16) * lda + half * 8;
  // B fragment (32x16 bf16): col n=tn*16+l16; K runs [half*8..+7] and [16+half*8..+7]
  const __bf16* brow = BT + (size_t)(tn * 16 + l16) * ldb + half * 8;

  v8f acc = {};
  for (int kk = k0; kk < kend; kk += 32) {
    v8bf a0 = *(const v8bf*)(arow + kk);
    v8bf a1 = *(const v8bf*)(arow + kk + 16);
    v8bf b0 = *(const v8bf*)(brow + kk);
    v8bf b1 = *(const v8bf*)(brow + kk + 16);
    v16bf a = __builtin_shufflevector(a0, a1, 0,1,2,3,4,5,6,7,8,9,10,11,12,13,14,15);
    v16bf b = __builtin_shufflevector(b0, b1, 0,1,2,3,4,5,6,7,8,9,10,11,12,13,14,15);
    acc = __builtin_amdgcn_wmma_f32_16x16x32_bf16(false, a, false, b,
                                                  (short)0, acc, false, false);
  }

  float* crow = C + (size_t)(tm * 16 + half * 8) * ldc + tn * 16 + l16;
  if (splitK > 1) {
#pragma unroll
    for (int r = 0; r < 8; ++r) unsafeAtomicAdd(crow + (size_t)r * ldc, acc[r]);
  } else {
    float bv = bias ? bias[tn * 16 + l16] : 0.0f;
#pragma unroll
    for (int r = 0; r < 8; ++r) crow[(size_t)r * ldc] = acc[r] + bv;
  }
}

// ---------------------------------------------------------------------------
// Packing kernels: f32 -> zero-padded bf16
// ---------------------------------------------------------------------------
// out[Mp,Kp] = in[M,K] (row-major copy + pad)
__global__ void k_cvt_pad(const float* __restrict__ in, __bf16* __restrict__ out,
                          int M, int K, int Mp, int Kp)
{
  long i = (long)blockIdx.x * blockDim.x + threadIdx.x;
  if (i >= (long)Mp * Kp) return;
  int m = (int)(i / Kp), k = (int)(i % Kp);
  float v = (m < M && k < K) ? in[(long)m * K + k] : 0.0f;
  out[i] = f2bf(v);
}

// out[Cp,Rp] = in[R,C]^T (LDS-tiled transpose + pad); blockDim (32,8)
__global__ void k_tcvt_pad(const float* __restrict__ in, __bf16* __restrict__ out,
                           int R, int Ccols, int Rp, int Cp)
{
  __shared__ float tile[32][33];
  int rb = blockIdx.y * 32, cb = blockIdx.x * 32;
  int tx = threadIdx.x, ty = threadIdx.y;
#pragma unroll
  for (int j = 0; j < 32; j += 8) {
    int r = rb + ty + j, c = cb + tx;
    tile[ty + j][tx] = (r < R && c < Ccols) ? in[(long)r * Ccols + c] : 0.0f;
  }
  __syncthreads();
#pragma unroll
  for (int j = 0; j < 32; j += 8) {
    int c = cb + ty + j, r = rb + tx;
    if (c < Cp && r < Rp) out[(long)c * Rp + r] = f2bf(tile[tx][ty + j]);
  }
}

// ---------------------------------------------------------------------------
// Elementwise / reduction helpers
// ---------------------------------------------------------------------------
__global__ void k_zero(float* __restrict__ p, long n) {
  long i = (long)blockIdx.x * blockDim.x + threadIdx.x;
  if (i < n) p[i] = 0.0f;
}

__global__ void k_init_gmax(float* g) { *g = -3.4e38f; }

// raw_feat = concat(lane(32), type(16), length(16), node(64))
__global__ void k_gather(const float* __restrict__ node_t, const float* __restrict__ type_t,
                         const float* __restrict__ len_t, const float* __restrict__ lane_t,
                         const int* __restrict__ nf, const int* __restrict__ tf,
                         const int* __restrict__ lf, const int* __restrict__ laf,
                         float* __restrict__ raw, int n)
{
  long i = (long)blockIdx.x * blockDim.x + threadIdx.x;
  if (i >= (long)n * 128) return;
  int row = (int)(i >> 7), c = (int)(i & 127);
  float v;
  if      (c < 32) v = lane_t[laf[row] * 32 + c];
  else if (c < 48) v = type_t[tf[row] * 16 + (c - 32)];
  else if (c < 64) v = len_t[lf[row] * 16 + (c - 48)];
  else             v = node_t[(long)nf[row] * 64 + (c - 64)];
  raw[i] = v;
}

// scale[c] = 1 / (relu(colsum(A[:,c]) - 1) + 1)
__global__ void k_col_scale(const float* __restrict__ A, float* __restrict__ scale,
                            int rows, int cols)
{
  __shared__ float red[256];
  int c = blockIdx.x;
  if (c >= cols) return;
  float s = 0.f;
  for (int r = threadIdx.x; r < rows; r += blockDim.x) s += A[(long)r * cols + c];
  red[threadIdx.x] = s; __syncthreads();
  for (int off = blockDim.x >> 1; off; off >>= 1) {
    if ((int)threadIdx.x < off) red[threadIdx.x] += red[threadIdx.x + off];
    __syncthreads();
  }
  if (threadIdx.x == 0) {
    float t = red[0] - 1.0f; t = t > 0.f ? t : 0.f;
    scale[c] = 1.0f / (t + 1.0f);
  }
}

// denomS[k] = relu(sum_j fnc_assign[k,j]*scale_f[j] - 1) + 1
__global__ void k_fa_denom(const float* __restrict__ fa, const float* __restrict__ scale_f,
                           float* __restrict__ denomS, int KSr, int KFr)
{
  int k = blockIdx.x * blockDim.x + threadIdx.x;
  if (k >= KSr) return;
  float s = 0.f;
  for (int j = 0; j < KFr; ++j) s += fa[(long)k * KFr + j] * scale_f[j];
  float t = s - 1.0f; t = t > 0.f ? t : 0.f;
  denomS[k] = t + 1.0f;
}

__global__ void k_rowscale(const float* __restrict__ in, const float* __restrict__ rs,
                           float* __restrict__ out, int M, int Ncols)
{
  long i = (long)blockIdx.x * blockDim.x + threadIdx.x;
  if (i >= (long)M * Ncols) return;
  out[i] = in[i] * rs[i / Ncols];
}

// ap = sigmoid(X) + 2*I   (fnc_adj = sigmoid+I, then _gcn adds another I)
__global__ void k_sigmoid2I(const float* __restrict__ X, float* __restrict__ out, int n) {
  long i = (long)blockIdx.x * blockDim.x + threadIdx.x;
  if (i >= (long)n * n) return;
  int r = (int)(i / n), c = (int)(i % n);
  float s = 1.0f / (1.0f + __expf(-X[i]));
  out[i] = s + ((r == c) ? 2.0f : 0.0f);
}

// sap = relu(struct_adj - 1e4*I) + 2*I
__global__ void k_sadj_post(const float* __restrict__ sa, float* __restrict__ out, int n) {
  long i = (long)blockIdx.x * blockDim.x + threadIdx.x;
  if (i >= (long)n * n) return;
  int r = (int)(i / n), c = (int)(i % n);
  float x = sa[i] - ((r == c) ? 1e4f : 0.0f);
  x = x > 0.f ? x : 0.f;
  out[i] = x + ((r == c) ? 2.0f : 0.0f);
}

// dinv[r] = rowsum(|A[r,:]|)^-1/2
__global__ void k_row_dinv(const float* __restrict__ A, float* __restrict__ dinv,
                           int M, int Kc, int ld)
{
  __shared__ float red[256];
  int r = blockIdx.x;
  if (r >= M) return;
  float s = 0.f;
  for (int j = threadIdx.x; j < Kc; j += blockDim.x) s += fabsf(A[(long)r * ld + j]);
  red[threadIdx.x] = s; __syncthreads();
  for (int off = blockDim.x >> 1; off; off >>= 1) {
    if ((int)threadIdx.x < off) red[threadIdx.x] += red[threadIdx.x + off];
    __syncthreads();
  }
  if (threadIdx.x == 0) dinv[r] = rsqrtf(red[0]);
}

__global__ void k_adjn(const float* __restrict__ A, const float* __restrict__ dinv,
                       float* __restrict__ out, int M, int Nc)
{
  long i = (long)blockIdx.x * blockDim.x + threadIdx.x;
  if (i >= (long)M * Nc) return;
  int r = (int)(i / Nc), c = (int)(i % Nc);
  out[i] = dinv[r] * A[i] * dinv[c];
}

// semb += 0.15 * fm / denomS[row]
__global__ void k_add_fm(float* __restrict__ y, const float* __restrict__ x,
                         const float* __restrict__ den, int M, int Ncols)
{
  long i = (long)blockIdx.x * blockDim.x + threadIdx.x;
  if (i >= (long)M * Ncols) return;
  y[i] += 0.15f * x[i] / den[i / Ncols];
}

__global__ void k_axpy(float* __restrict__ y, const float* __restrict__ x, float a, long n) {
  long i = (long)blockIdx.x * blockDim.x + threadIdx.x;
  if (i < n) y[i] += a * x[i];
}

// hs[r] = h[r,:]·ag[0:128] ; hd[r] = h[r,:]·ag[128:256]   (wave per row)
__global__ void k_attn_dots(const float* __restrict__ h, const float* __restrict__ ag,
                            float* __restrict__ hs, float* __restrict__ hd, int n)
{
  int row = blockIdx.x * (blockDim.x >> 5) + (threadIdx.x >> 5);
  if (row >= n) return;
  int lane = threadIdx.x & 31;
  const float* r = h + (long)row * 128;
  float s0 = 0.f, s1 = 0.f;
  for (int j = lane; j < 128; j += 32) {
    float x = r[j];
    s0 += x * ag[j];
    s1 += x * ag[128 + j];
  }
  for (int off = 16; off; off >>= 1) {
    s0 += __shfl_down(s0, off, 32);
    s1 += __shfl_down(s1, off, 32);
  }
  if (lane == 0) { hs[row] = s0; hd[row] = s1; }
}

// vals[e] = leakyrelu(hs[src]+hd[dst]); track global max
__global__ void k_edge_vals(const int* __restrict__ ei, const float* __restrict__ hs,
                            const float* __restrict__ hd, float* __restrict__ vals,
                            float* __restrict__ gmax, int E)
{
  __shared__ float red[256];
  int e = blockIdx.x * blockDim.x + threadIdx.x;
  float v = -3.4e38f;
  if (e < E) {
    float x = hs[ei[e]] + hd[ei[E + e]];
    x = x > 0.f ? x : 0.2f * x;
    vals[e] = x;
    v = x;
  }
  red[threadIdx.x] = v; __syncthreads();
  for (int off = blockDim.x >> 1; off; off >>= 1) {
    if ((int)threadIdx.x < off) red[threadIdx.x] = fmaxf(red[threadIdx.x], red[threadIdx.x + off]);
    __syncthreads();
  }
  if (threadIdx.x == 0) atomicMaxF32(gmax, red[0]);
}

// wave per edge: hp[src,:] += ev * h[dst,:] ; rowsum[src] += ev
__global__ void k_edge_scatter(const int* __restrict__ ei, const float* __restrict__ vals,
                               const float* __restrict__ gmax, const float* __restrict__ h,
                               float* __restrict__ hp, float* __restrict__ rowsum, int E)
{
  int e = blockIdx.x * (blockDim.x >> 5) + (threadIdx.x >> 5);
  if (e >= E) return;
  int lane = threadIdx.x & 31;
  int s = ei[e], d = ei[E + e];
  float ev = __expf(vals[e] - *gmax);
  const float* hrow = h + (long)d * 128;
  float* orow = hp + (long)s * 128;
#pragma unroll
  for (int j = 0; j < 4; ++j) {
    int c = lane * 4 + j;
    unsafeAtomicAdd(&orow[c], ev * hrow[c]);
  }
  if (lane == 0) unsafeAtomicAdd(&rowsum[s], ev);
}

// out = elu(hp / (rowsum + 1e-15))
__global__ void k_gat_final(const float* __restrict__ hp, const float* __restrict__ rowsum,
                            float* __restrict__ out, int n)
{
  long i = (long)blockIdx.x * blockDim.x + threadIdx.x;
  if (i >= (long)n * 128) return;
  float x = hp[i] / (rowsum[i >> 7] + 1e-15f);
  out[i] = x > 0.f ? x : (__expf(x) - 1.0f);
}

// ---------------------------------------------------------------------------
// Host orchestration
// ---------------------------------------------------------------------------
extern "C" void kernel_launch(void* const* d_in, const int* in_sizes, int n_in,
                              void* d_out, int out_size, void* d_ws, size_t ws_size,
                              hipStream_t stream)
{
  (void)in_sizes; (void)n_in; (void)out_size; (void)ws_size;
  constexpr int N = 20000, D = 128, KS = 500, KSP = 512, KF = 64, E = 640000;

  const float* node_table   = (const float*)d_in[0];
  const float* type_table   = (const float*)d_in[1];
  const float* length_table = (const float*)d_in[2];
  const float* lane_table   = (const float*)d_in[3];
  const float* struct_assign= (const float*)d_in[4];
  const float* fnc_assign   = (const float*)d_in[5];
  const float* struct_adj   = (const float*)d_in[6];
  const float* Wf[2] = {(const float*)d_in[7],  (const float*)d_in[13]};
  const float* bf[2] = {(const float*)d_in[8],  (const float*)d_in[14]};
  const float* Ws[2] = {(const float*)d_in[9],  (const float*)d_in[15]};
  const float* bs[2] = {(const float*)d_in[10], (const float*)d_in[16]};
  const float* Wg[2] = {(const float*)d_in[11], (const float*)d_in[17]};
  const float* ag[2] = {(const float*)d_in[12], (const float*)d_in[18]};
  const int* nf  = (const int*)d_in[19];
  const int* tf  = (const int*)d_in[20];
  const int* lf  = (const int*)d_in[21];
  const int* laf = (const int*)d_in[22];
  const int* ei  = (const int*)d_in[23];
  float* out = (float*)d_out;

  // ---- workspace carve ----
  char* p = (char*)d_ws;
  auto carve = [&](size_t bytes) { void* r = (void*)p; p += (bytes + 255) & ~(size_t)255; return r; };
  // f32 buffers
  float* raw    = (float*)carve((size_t)N * D * 4);
  float* h      = (float*)carve((size_t)N * D * 4);
  float* smhp   = (float*)carve((size_t)N * D * 4);   // struct_message, then GAT hp
  float* vals   = (float*)carve((size_t)E * 4);
  float* hs     = (float*)carve((size_t)N * 4);
  float* hd     = (float*)carve((size_t)N * 4);
  float* rowsum = (float*)carve((size_t)N * 4);
  float* C1     = (float*)carve((size_t)KSP * D * 4);
  float* semb   = (float*)carve((size_t)KSP * D * 4);
  float* fm     = (float*)carve((size_t)KSP * D * 4);
  float* ssup   = (float*)carve((size_t)KSP * D * 4);
  float* feP    = (float*)carve((size_t)KF * D * 4);
  float* fe0    = (float*)carve((size_t)KF * D * 4);
  float* fe1    = (float*)carve((size_t)KF * D * 4);
  float* fsup   = (float*)carve((size_t)KF * D * 4);
  float* fadjC  = (float*)carve((size_t)KF * KF * 4);
  float* fap    = (float*)carve((size_t)KF * KF * 4);
  float* fadjn  = (float*)carve((size_t)KF * KF * 4);
  float* sap    = (float*)carve((size_t)KS * KS * 4);
  float* sadjn  = (float*)carve((size_t)KS * KS * 4);
  float* scale_s= (float*)carve(KSP * 4);
  float* scale_f= (float*)carve(KF * 4);
  float* denomS = (float*)carve(KSP * 4);
  float* dinv_f = (float*)carve(KF * 4);
  float* dinv_s = (float*)carve(KSP * 4);
  float* gmax   = (float*)carve(4);
  // packed bf16 operands (static: packed once, reused by both passes)
  __bf16* saT_bf  = (__bf16*)carve((size_t)KSP * N * 2);   // [512,20000] = struct_assign^T
  __bf16* sa_bf   = (__bf16*)carve((size_t)N * KSP * 2);   // [20000,512] = struct_assign
  __bf16* faT_bf  = (__bf16*)carve((size_t)KF * KSP * 2);  // [64,512]  = fnc_assign^T
  __bf16* fa_bf   = (__bf16*)carve((size_t)KSP * KF * 2);  // [512,64]  = fnc_assign
  __bf16* WfT_bf[2], *WsT_bf[2], *WgT_bf[2];
  for (int l = 0; l < 2; ++l) {
    WfT_bf[l] = (__bf16*)carve((size_t)D * D * 2);
    WsT_bf[l] = (__bf16*)carve((size_t)D * D * 2);
    WgT_bf[l] = (__bf16*)carve((size_t)D * D * 2);
  }
  // packed bf16 operands (per-pass, reused)
  __bf16* raw_bf   = (__bf16*)carve((size_t)N * D * 2);    // [20000,128]
  __bf16* rawT_bf  = (__bf16*)carve((size_t)D * N * 2);    // [128,20000]
  __bf16* semb_bf  = (__bf16*)carve((size_t)KSP * D * 2);  // [512,128]
  __bf16* sembT_bf = (__bf16*)carve((size_t)D * KSP * 2);  // [128,512]
  __bf16* fe0_bf   = (__bf16*)carve((size_t)KF * D * 2);   // [64,128]
  __bf16* fsupT_bf = (__bf16*)carve((size_t)D * KF * 2);   // [128,64]
  __bf16* fadjn_bf = (__bf16*)carve((size_t)KF * KF * 2);  // [64,64]
  __bf16* fe1T_bf  = (__bf16*)carve((size_t)D * KF * 2);   // [128,64]
  __bf16* sadjn_bf = (__bf16*)carve((size_t)KSP * KSP * 2);// [512,512]
  __bf16* ssupT_bf = (__bf16*)carve((size_t)D * KSP * 2);  // [128,512]
  __bf16* c1T_bf   = (__bf16*)carve((size_t)D * KSP * 2);  // [128,512]

  auto blocks1d = [](long n, int bd) { return (unsigned)((n + bd - 1) / bd); };

  auto gemm = [&](const __bf16* A, const __bf16* BT, float* C, const float* bias,
                  int Mp, int Np, int Kp, int lda, int ldb, int ldc, int splitK) {
    int tiles = (Mp >> 4) * (Np >> 4);
    dim3 grid((tiles + 7) / 8, splitK);
    k_wmma_gemm<<<grid, 256, 0, stream>>>(A, BT, C, bias, Mp, Np, Kp, lda, ldb, ldc, splitK);
  };
  auto cvt = [&](const float* in, __bf16* o, int M, int K, int Mp, int Kp) {
    k_cvt_pad<<<blocks1d((long)Mp * Kp, 256), 256, 0, stream>>>(in, o, M, K, Mp, Kp);
  };
  auto tcvt = [&](const float* in, __bf16* o, int R, int C, int Rp, int Cp) {
    dim3 grid((Cp + 31) / 32, (Rp + 31) / 32), blk(32, 8);
    k_tcvt_pad<<<grid, blk, 0, stream>>>(in, o, R, C, Rp, Cp);
  };

  // ---- setup (once per call) ----
  k_gather<<<blocks1d((long)N * D, 256), 256, 0, stream>>>(
      node_table, type_table, length_table, lane_table, nf, tf, lf, laf, raw, N);
  k_col_scale<<<KS, 256, 0, stream>>>(struct_assign, scale_s, N, KS);
  k_col_scale<<<KF, 256, 0, stream>>>(fnc_assign, scale_f, KS, KF);
  k_fa_denom<<<blocks1d(KS, 256), 256, 0, stream>>>(fnc_assign, scale_f, denomS, KS, KF);
  tcvt(struct_assign, saT_bf, N, KS, N, KSP);      // A: [512,20000]
  cvt (struct_assign, sa_bf,  N, KS, N, KSP);      // A: [20000,512]
  tcvt(fnc_assign, faT_bf, KS, KF, KSP, KF);       // A: [64,512]
  cvt (fnc_assign, fa_bf,  KS, KF, KSP, KF);       // A: [512,64]
  for (int l = 0; l < 2; ++l) {                    // BT for x @ W GEMMs
    tcvt(Wf[l], WfT_bf[l], D, D, D, D);
    tcvt(Ws[l], WsT_bf[l], D, D, D, D);
    tcvt(Wg[l], WgT_bf[l], D, D, D, D);
  }

  // ---- two core passes ----
  for (int pass = 0; pass < 2; ++pass) {
    // struct_emb = diag(scale_s) * (struct_assign^T @ raw)   [split-K=16]
    tcvt(raw, rawT_bf, N, D, N, D);                                  // BT [128,20000]
    k_zero<<<blocks1d((long)KSP * D, 256), 256, 0, stream>>>(C1, (long)KSP * D);
    gemm(saT_bf, rawT_bf, C1, nullptr, KSP, D, N, N, N, D, 16);
    k_rowscale<<<blocks1d((long)KS * D, 256), 256, 0, stream>>>(C1, scale_s, semb, KS, D);

    // fnc_emb0 = diag(scale_f) * (fnc_assign^T @ struct_emb)
    tcvt(semb, sembT_bf, KS, D, KSP, D);                             // BT [128,512]
    gemm(faT_bf, sembT_bf, feP, nullptr, KF, D, KSP, KSP, KSP, D, 1);
    k_rowscale<<<blocks1d((long)KF * D, 256), 256, 0, stream>>>(feP, scale_f, fe0, KF, D);
    cvt(fe0, fe0_bf, KF, D, KF, D);

    // fnc adjacency: sigmoid(fe0 @ fe0^T) + 2I, symmetric-normalize
    gemm(fe0_bf, fe0_bf, fadjC, nullptr, KF, KF, D, D, D, KF, 1);    // BT of fe0^T is fe0
    k_sigmoid2I<<<blocks1d((long)KF * KF, 256), 256, 0, stream>>>(fadjC, fap, KF);
    k_row_dinv<<<KF, 256, 0, stream>>>(fap, dinv_f, KF, KF, KF);
    k_adjn<<<blocks1d((long)KF * KF, 256), 256, 0, stream>>>(fap, dinv_f, fadjn, KF, KF);
    cvt(fadjn, fadjn_bf, KF, KF, KF, KF);

    // fnc GCN: fe1 = adjn @ (fe0 @ Wf) + bf
    gemm(fe0_bf, WfT_bf[pass], fsup, nullptr, KF, D, D, D, D, D, 1);
    tcvt(fsup, fsupT_bf, KF, D, KF, D);                              // BT [128,64]
    gemm(fadjn_bf, fsupT_bf, fe1, bf[pass], KF, D, KF, KF, KF, D, 1);

    // struct_emb += 0.15 * (fnc_assign @ fe1) / denomS
    tcvt(fe1, fe1T_bf, KF, D, KF, D);                                // BT [128,64]
    gemm(fa_bf, fe1T_bf, fm, nullptr, KSP, D, KF, KF, KF, D, 1);
    k_add_fm<<<blocks1d((long)KS * D, 256), 256, 0, stream>>>(semb, fm, denomS, KS, D);

    // struct GCN adjacency
    k_sadj_post<<<blocks1d((long)KS * KS, 256), 256, 0, stream>>>(struct_adj, sap, KS);
    k_row_dinv<<<KS, 256, 0, stream>>>(sap, dinv_s, KS, KS, KS);
    k_adjn<<<blocks1d((long)KS * KS, 256), 256, 0, stream>>>(sap, dinv_s, sadjn, KS, KS);
    cvt(sadjn, sadjn_bf, KS, KS, KSP, KSP);                          // A [512,512]

    // struct GCN: semb' = sadjn @ (semb @ Ws) + bs
    cvt(semb, semb_bf, KS, D, KSP, D);                               // A [512,128]
    gemm(semb_bf, WsT_bf[pass], ssup, nullptr, KSP, D, D, D, D, D, 1);
    tcvt(ssup, ssupT_bf, KS, D, KSP, D);                             // BT [128,512]
    gemm(sadjn_bf, ssupT_bf, C1, bs[pass], KSP, D, KSP, KSP, KSP, D, 1);

    // raw += 0.5 * (struct_assign @ semb')
    tcvt(C1, c1T_bf, KS, D, KSP, D);                                 // BT [128,512]
    gemm(sa_bf, c1T_bf, smhp, nullptr, N, D, KSP, KSP, KSP, D, 1);
    k_axpy<<<blocks1d((long)N * D, 256), 256, 0, stream>>>(raw, smhp, 0.5f, (long)N * D);

    // ---- spGAT ----
    cvt(raw, raw_bf, N, D, N, D);                                    // A [20000,128]
    gemm(raw_bf, WgT_bf[pass], h, nullptr, N, D, D, D, D, D, 1);
    k_attn_dots<<<blocks1d((long)N, 8), 256, 0, stream>>>(h, ag[pass], hs, hd, N);
    k_init_gmax<<<1, 1, 0, stream>>>(gmax);
    k_edge_vals<<<blocks1d((long)E, 256), 256, 0, stream>>>(ei, hs, hd, vals, gmax, E);
    k_zero<<<blocks1d((long)N * D, 256), 256, 0, stream>>>(smhp, (long)N * D);
    k_zero<<<blocks1d((long)N, 256), 256, 0, stream>>>(rowsum, N);
    k_edge_scatter<<<blocks1d((long)E, 8), 256, 0, stream>>>(ei, vals, gmax, h, smhp, rowsum, E);
    float* dest = (pass == 0) ? raw : out;
    k_gat_final<<<blocks1d((long)N * D, 256), 256, 0, stream>>>(smhp, rowsum, dest, N);
  }
}
